// ScaledBlock_7722351199078
// MI455X (gfx1250) — compile-verified, matrix operand order checked by reading, not run on hardware
//
#include <hip/hip_runtime.h>

// ---------------------------------------------------------------------------
// Types for WMMA intrinsics (CDNA5 / gfx1250, wave32)
// ---------------------------------------------------------------------------
typedef __bf16 v16bf __attribute__((ext_vector_type(16)));
typedef float  v8f   __attribute__((ext_vector_type(8)));

union ABPack {
    uint4 q[2];   // 32 bytes = 8 VGPRs
    v16bf v;      // 16 bf16 elements
};

// f32 -> bf16 with round-to-nearest-even
__device__ __forceinline__ unsigned short f2bf(float f) {
    union { float f; unsigned int u; } x;
    x.f = f;
    unsigned int r = x.u + 0x7FFFu + ((x.u >> 16) & 1u);
    return (unsigned short)(r >> 16);
}

// ---------------------------------------------------------------------------
// Weight pre-swizzle: W[set][27][64][64] f32 row-major (in,out) ->
// wsw[set][k][ntile][chunk][lane][v] packed bf16 pairs, matching the
// 16-bit B-matrix (32x16 per WMMA) VGPR layout:
//   lane in [0,15]  -> N = ntile*16 + lane,    K-halves {0..7, 16..23} (+32*chunk)
//   lane in [16,31] -> N = ntile*16 + lane-16, K-halves {8..15, 24..31}
// ---------------------------------------------------------------------------
__global__ void prep_weights_kernel(const float* __restrict__ W0,
                                    const float* __restrict__ W1,
                                    const float* __restrict__ W2,
                                    const float* __restrict__ W3,
                                    unsigned int* __restrict__ wsw) {
    const int total = 4 * 27 * 4 * 2 * 32 * 8;
    int id = blockIdx.x * blockDim.x + threadIdx.x;
    if (id >= total) return;

    int v    = id & 7;
    int lane = (id >> 3) & 31;
    int c    = (id >> 8) & 1;
    int nt   = (id >> 9) & 3;
    int rest = id >> 11;          // set*27 + k
    int k    = rest % 27;
    int set  = rest / 27;

    const float* W = (set == 0) ? W0 : (set == 1) ? W1 : (set == 2) ? W2 : W3;

    int g    = lane >> 4;
    int kl   = (v < 4) ? (8 * g + 2 * v) : (16 + 8 * g + 2 * (v - 4));
    int krow = 32 * c + kl;
    int n    = nt * 16 + (lane & 15);

    float f0 = W[(k * 64 + krow) * 64 + n];
    float f1 = W[(k * 64 + krow + 1) * 64 + n];
    wsw[id] = (unsigned int)f2bf(f0) | ((unsigned int)f2bf(f1) << 16);
}

// ---------------------------------------------------------------------------
// feats f32 -> bf16
// ---------------------------------------------------------------------------
__global__ void f32_to_bf16_kernel(const float* __restrict__ in,
                                   unsigned short* __restrict__ out, int n) {
    int id = blockIdx.x * blockDim.x + threadIdx.x;
    if (id < n) out[id] = f2bf(in[id]);
}

// ---------------------------------------------------------------------------
// Sparse conv: one block = 128 threads = 4 waves, 64-point tile.
// Wave w owns output columns [16w, 16w+16) for all 4 M-tiles (64 points).
// Per k: 4 x b128 B loads + 16 x ds_load_b128 A loads grouped AHEAD of the
// 8 WMMAs via sched_group_barrier so dscnt waits stagger instead of draining
// before every WMMA. k+1 gather prefetched before compute; double-buffered
// LDS A-tile; one __syncthreads per k.
// MODE: 0 = bias+relu->bf16, 1 = bias+FiLM->bf16,
//       2 = bias+relu->bf16, 3 = bias+relu+residual->f32
// ---------------------------------------------------------------------------
template <int MODE>
__global__ __launch_bounds__(128)
void sparse_conv_kernel(const unsigned short* __restrict__ xbf,  // [P][64] bf16
                        const int* __restrict__ nbr,             // [P][27]
                        const unsigned int* __restrict__ wsw,    // swizzled W (this set)
                        const float* __restrict__ bias,          // [64]
                        const float* __restrict__ extra,         // cond (MODE1) / feats (MODE3)
                        unsigned short* __restrict__ out_bf,     // MODE 0/1/2
                        float* __restrict__ out_f32,             // MODE 3
                        int P) {
    __shared__ alignas(16) unsigned int As[2][64 * 32];  // 2 x 64 rows x 64 bf16 = 16KB

    const int t    = threadIdx.x;
    const int lane = t & 31;
    const int wv   = t >> 5;        // n-tile 0..3
    const int g    = lane >> 4;     // lane half
    const int m    = lane & 15;     // A row within 16-tile, B/D column within 16-tile
    const int p0   = blockIdx.x * 64;

    // gather role: 2 threads per row, 4 quads (64B) each
    const int gr   = t >> 1;        // row 0..63
    const int gq   = (t & 1) * 4;   // first quad within row (0 or 4)
    const int prow = p0 + gr;

    v8f acc[4];
#pragma unroll
    for (int mt = 0; mt < 4; ++mt)
#pragma unroll
        for (int i = 0; i < 8; ++i) acc[mt][i] = 0.f;

    // ---- prologue: gather k = 0 into buffer 0 ----
    {
        int idx = (prow < P) ? nbr[prow * 27 + 0] : -1;
        uint4 z4; z4.x = z4.y = z4.z = z4.w = 0u;
        uint4 nv[4] = {z4, z4, z4, z4};
        if (idx >= 0) {
            const uint4* src = (const uint4*)(xbf + (size_t)idx * 64);
#pragma unroll
            for (int j = 0; j < 4; ++j) nv[j] = src[gq + j];
        }
#pragma unroll
        for (int j = 0; j < 4; ++j)
            *(uint4*)&As[0][gr * 32 + (gq + j) * 4] = nv[j];
    }

    for (int k = 0; k < 27; ++k) {
        const int cur = k & 1;
        __syncthreads();  // tile k visible; tile (k-1) reads retired

        // ---- issue gather loads for k+1 (latency hidden behind WMMAs) ----
        uint4 z4; z4.x = z4.y = z4.z = z4.w = 0u;
        uint4 nv[4] = {z4, z4, z4, z4};
        const bool have_next = (k + 1 < 27);
        if (have_next) {
            int nidx = (prow < P) ? nbr[prow * 27 + (k + 1)] : -1;
            if (nidx >= 0) {
                const uint4* src = (const uint4*)(xbf + (size_t)nidx * 64);
#pragma unroll
                for (int j = 0; j < 4; ++j) nv[j] = src[gq + j];
            }
        }

        // ---- issue ALL operand loads for tile k up front ----
        const uint4* wp =
            (const uint4*)(wsw + (((k * 4 + wv) * 2) * 32 + lane) * 8);
        ABPack b[2];
        b[0].q[0] = wp[0];                 // chunk 0: K = 0..31
        b[0].q[1] = wp[1];
        b[1].q[0] = wp[64];                // chunk 1: K = 32..63 (+32 lanes * 8 dwords)
        b[1].q[1] = wp[65];

        const unsigned int* abase = &As[cur][0];
        ABPack a[2][4];
#pragma unroll
        for (int c = 0; c < 2; ++c)
#pragma unroll
            for (int mt = 0; mt < 4; ++mt) {
                const unsigned int* ap = abase + (mt * 16 + m) * 32 + c * 16 + g * 4;
                a[c][mt].q[0] = *(const uint4*)(ap);
                a[c][mt].q[1] = *(const uint4*)(ap + 8);
            }

        // ---- 8 back-to-back WMMAs (chain only on accumulators) ----
#pragma unroll
        for (int c = 0; c < 2; ++c)
#pragma unroll
            for (int mt = 0; mt < 4; ++mt)
                acc[mt] = __builtin_amdgcn_wmma_f32_16x16x32_bf16(
                    false, a[c][mt].v, false, b[c].v, (short)0, acc[mt], false, false);

        // ---- commit gathered tile k+1 into the other buffer ----
        if (have_next) {
#pragma unroll
            for (int j = 0; j < 4; ++j)
                *(uint4*)&As[cur ^ 1][gr * 32 + (gq + j) * 4] = nv[j];
        }

        // Scheduler pinning for this iteration's compute block:
        //   all 16 DS reads (A tiles) first, then the 8 WMMAs, then DS writes.
        // This forces distinct A registers -> staggered s_wait_dscnt instead of
        // a full dscnt drain before every WMMA.
        __builtin_amdgcn_sched_group_barrier(0x100, 16, 0);  // DS reads
        __builtin_amdgcn_sched_group_barrier(0x008, 8, 0);   // MFMA/WMMA
        __builtin_amdgcn_sched_group_barrier(0x200, 4, 0);   // DS writes
    }

    // ---- epilogue: D layout -> (row = mt*16 + r + 8g, col n = wv*16 + m) ----
    const int n  = wv * 16 + m;
    const float bn = bias[n];
#pragma unroll
    for (int mt = 0; mt < 4; ++mt) {
#pragma unroll
        for (int r = 0; r < 8; ++r) {
            int p = p0 + mt * 16 + r + 8 * g;
            if (p >= P) continue;
            float v = acc[mt][r] + bn;
            if (MODE == 0 || MODE == 2) {
                v = fmaxf(v, 0.f);
                out_bf[(size_t)p * 64 + n] = f2bf(v);
            } else if (MODE == 1) {
                float beta  = extra[(size_t)p * 128 + n];
                float gamma = extra[(size_t)p * 128 + 64 + n];
                v = v * beta + gamma;
                out_bf[(size_t)p * 64 + n] = f2bf(v);
            } else {  // MODE == 3
                v = fmaxf(v, 0.f) + extra[(size_t)p * 64 + n];
                out_f32[(size_t)p * 64 + n] = v;
            }
        }
    }
}

// ---------------------------------------------------------------------------
// Host launcher
// ---------------------------------------------------------------------------
extern "C" void kernel_launch(void* const* d_in, const int* in_sizes, int n_in,
                              void* d_out, int out_size, void* d_ws, size_t ws_size,
                              hipStream_t stream) {
    const float* feats = (const float*)d_in[0];
    const float* cond  = (const float*)d_in[1];
    const float* W1a   = (const float*)d_in[2];
    const float* b1a   = (const float*)d_in[3];
    const float* W1b   = (const float*)d_in[4];
    const float* b1b   = (const float*)d_in[5];
    const float* W2a   = (const float*)d_in[6];
    const float* b2a   = (const float*)d_in[7];
    const float* W2b   = (const float*)d_in[8];
    const float* b2b   = (const float*)d_in[9];
    const int*   nbr   = (const int*)d_in[10];

    const int P = in_sizes[0] / 64;

    // workspace layout: [swizzled weights: 4 sets][bufA bf16][bufB bf16]
    const size_t kWswDwordsPerSet = 27 * 4 * 2 * 32 * 8;  // 55296
    unsigned int* wsw = (unsigned int*)d_ws;
    size_t wswBytes = 4 * kWswDwordsPerSet * sizeof(unsigned int);
    size_t off = (wswBytes + 255) & ~(size_t)255;
    size_t bufBytes = ((size_t)P * 64 * sizeof(unsigned short) + 255) & ~(size_t)255;
    unsigned short* bufA = (unsigned short*)((char*)d_ws + off);
    unsigned short* bufB = (unsigned short*)((char*)d_ws + off + bufBytes);

    {
        int tot = 4 * (int)kWswDwordsPerSet;
        prep_weights_kernel<<<(tot + 255) / 256, 256, 0, stream>>>(W1a, W1b, W2a, W2b, wsw);
    }
    {
        int tot = P * 64;
        f32_to_bf16_kernel<<<(tot + 255) / 256, 256, 0, stream>>>(feats, bufA, tot);
    }

    const int nTiles = (P + 63) / 64;
    sparse_conv_kernel<0><<<nTiles, 128, 0, stream>>>(
        bufA, nbr, wsw + 0 * kWswDwordsPerSet, b1a, nullptr, bufB, nullptr, P);
    sparse_conv_kernel<1><<<nTiles, 128, 0, stream>>>(
        bufB, nbr, wsw + 1 * kWswDwordsPerSet, b1b, cond, bufA, nullptr, P);
    sparse_conv_kernel<2><<<nTiles, 128, 0, stream>>>(
        bufA, nbr, wsw + 2 * kWswDwordsPerSet, b2a, nullptr, bufB, nullptr, P);
    sparse_conv_kernel<3><<<nTiles, 128, 0, stream>>>(
        bufB, nbr, wsw + 3 * kWswDwordsPerSet, b2b, feats, nullptr, (float*)d_out, P);
}
